// GAT_64974265254098
// MI455X (gfx1250) — matrix operand order repaired and checked by reading, not compiled, not run on hardware
//
#include <hip/hip_runtime.h>
#include <hip/hip_fp16.h>

typedef __attribute__((ext_vector_type(16))) _Float16 v16h;
typedef __attribute__((ext_vector_type(2)))  _Float16 v2h;
typedef __attribute__((ext_vector_type(8)))  float    v8f;

#define NEG_SLOPE 0.2f

// ---------------------------------------------------------------------------
// fp32 -> fp16 pre-conversion (paired): out[2i..2i+1] = (f16)in[2i..2i+1]
// ---------------------------------------------------------------------------
__global__ void cvt_f32_to_f16(const float* __restrict__ in,
                               _Float16* __restrict__ out, int n2)
{
    const int i = blockIdx.x * blockDim.x + threadIdx.x;
    if (i >= n2) return;
    const float2 v = ((const float2*)in)[i];
    v2h p;
    p.x = (_Float16)v.x;
    p.y = (_Float16)v.y;
    ((v2h*)out)[i] = p;
}

// ---------------------------------------------------------------------------
// GEMM: feat[n, OUT_D] = X16[n, IN_D] @ W16[OUT_D, IN_D]^T, f32 accumulation.
// One wave32 computes a full 16 x OUT_D row-strip: the A fragment is loaded
// once per K-block and reused across all OUT_D/16 WMMA tiles.
// Fully unrolled -> IN_D/32 * OUT_D/16 v_wmma_f32_16x16x32_f16 per kernel.
// ---------------------------------------------------------------------------
template<int IN_D, int OUT_D>
__global__ __launch_bounds__(128) void gat_gemm_wmma(
    const _Float16* __restrict__ X16, const _Float16* __restrict__ W16,
    float* __restrict__ feat, int nrows)
{
    constexpr int NT = OUT_D / 16;
    const int lane   = threadIdx.x;          // 0..31
    const int mtiles = (nrows + 15) >> 4;
    const int mtile  = blockIdx.x * blockDim.y + threadIdx.y;
    if (mtile >= mtiles) return;             // wave-uniform -> EXEC all-1s

    const int half  = lane & 15;
    const int phase = lane >> 4;             // 0 or 1

    int row = mtile * 16 + half;             // A row held by this lane
    const bool full = (mtile * 16 + 16) <= nrows;   // wave-uniform
    if (row >= nrows) row = nrows - 1;       // clamp loads only
    const _Float16* __restrict__ xr = X16 + (size_t)row * IN_D;

    v8f acc[NT];
#pragma unroll
    for (int t = 0; t < NT; ++t) acc[t] = (v8f){};

#pragma unroll
    for (int kb = 0; kb < IN_D; kb += 32) {
        // A fragment: VGPR v holds K = 2v + 16*phase (+pair)
        v16h a;
#pragma unroll
        for (int v = 0; v < 8; ++v) {
            const int k0 = kb + 2 * v + phase * 16;
            const v2h p = *(const v2h*)(xr + k0);
            a[2 * v]     = p.x;
            a[2 * v + 1] = p.y;
        }
        // B fragment per N-tile: element e holds K = kb + 16*phase + e,
        // i.e. 16 contiguous halves (32 B) of one W row -> 2x b128 loads.
#pragma unroll
        for (int t = 0; t < NT; ++t) {
            const v16h b = *(const v16h*)(W16 + (size_t)(t * 16 + half) * IN_D
                                          + kb + 16 * phase);
            acc[t] = __builtin_amdgcn_wmma_f32_16x16x32_f16(
                false, a, false, b, (short)0, acc[t], false, false);
        }
    }

    // C/D layout: lane holds column (lane&15); VGPR r holds row r + 8*phase
    if (full) {
#pragma unroll
        for (int t = 0; t < NT; ++t) {
            float* op = feat + (size_t)(mtile * 16 + 8 * phase) * OUT_D
                        + t * 16 + half;
#pragma unroll
            for (int r = 0; r < 8; ++r)
                op[(size_t)r * OUT_D] = acc[t][r];
        }
    } else {
#pragma unroll
        for (int t = 0; t < NT; ++t)
            for (int r = 0; r < 8; ++r) {
                const int orow = mtile * 16 + 8 * phase + r;
                if (orow < nrows)
                    feat[(size_t)orow * OUT_D + t * 16 + half] = acc[t][r];
            }
    }
}

// ---------------------------------------------------------------------------
// Attention dots: el[n,h] = <feat[n,h,:], al[h,:]> ; er likewise.
// ---------------------------------------------------------------------------
__global__ void gat_el_er(const float* __restrict__ feat,
                          const float* __restrict__ al,
                          const float* __restrict__ ar,
                          float* __restrict__ el, float* __restrict__ er,
                          int n, int h, int dh)
{
    const int i = blockIdx.x * blockDim.x + threadIdx.x;
    if (i >= n * h) return;
    const int node = i / h, hh = i % h;
    const float* f = feat + (size_t)node * h * dh + (size_t)hh * dh;
    const float* a = al + hh * dh;
    const float* r = ar + hh * dh;
    float sl = 0.f, sr = 0.f;
    for (int j = 0; j < dh; ++j) { const float v = f[j]; sl += v * a[j]; sr += v * r[j]; }
    el[i] = sl; er[i] = sr;
}

// ---------------------------------------------------------------------------
// Per-layer init: agg = 0, segment-max = -inf, segment-sum = 0.
// ---------------------------------------------------------------------------
__global__ void gat_init(float* __restrict__ mmax, float* __restrict__ denom,
                         float* __restrict__ agg, int n, int h, int out_d)
{
    const int i = blockIdx.x * blockDim.x + threadIdx.x;
    if (i < n * out_d) agg[i] = 0.f;
    if (i < n * h) { mmax[i] = -3.402823e38f; denom[i] = 0.f; }
}

__device__ __forceinline__ void atomicMaxF(float* addr, float val)
{
    unsigned int* p = (unsigned int*)addr;
    unsigned int old = *p;
    while (__uint_as_float(old) < val) {
        const unsigned int assumed = old;
        old = atomicCAS(p, assumed, __float_as_uint(val));
        if (old == assumed) break;
    }
}

__device__ __forceinline__ float leaky(float v)
{
    return (v > 0.f) ? v : NEG_SLOPE * v;
}

// ---------------------------------------------------------------------------
// Edge softmax pass 1: segment max over destinations.
// ---------------------------------------------------------------------------
__global__ void gat_edge_max(const int* __restrict__ src, const int* __restrict__ dst,
                             const float* __restrict__ el, const float* __restrict__ er,
                             float* __restrict__ mmax, int E, int h)
{
    const int i = blockIdx.x * blockDim.x + threadIdx.x;
    if (i >= E * h) return;
    const int e = i / h, hh = i % h;
    const float v = leaky(el[src[e] * h + hh] + er[dst[e] * h + hh]);
    atomicMaxF(&mmax[dst[e] * h + hh], v);
}

// ---------------------------------------------------------------------------
// Edge softmax pass 2: ex = exp(e - max[dst]); denom[dst] += ex.
// ---------------------------------------------------------------------------
__global__ void gat_edge_expsum(const int* __restrict__ src, const int* __restrict__ dst,
                                const float* __restrict__ el, const float* __restrict__ er,
                                const float* __restrict__ mmax,
                                float* __restrict__ exb, float* __restrict__ denom,
                                int E, int h)
{
    const int i = blockIdx.x * blockDim.x + threadIdx.x;
    if (i >= E * h) return;
    const int e = i / h, hh = i % h;
    const int d = dst[e];
    const float v  = leaky(el[src[e] * h + hh] + er[d * h + hh]);
    const float ex = __expf(v - mmax[d * h + hh]);
    exb[i] = ex;
    atomicAdd(&denom[d * h + hh], ex);
}

// ---------------------------------------------------------------------------
// Aggregation: agg[dst] += alpha * feat[src].  One thread per (edge, float4).
// feat (<=25.6 MB) is L2-resident on MI455X (192 MB L2), so the random
// gathers and f32 atomics never leave the chip.
// ---------------------------------------------------------------------------
__global__ void gat_edge_aggregate(const int* __restrict__ src, const int* __restrict__ dst,
                                   const float* __restrict__ exb,
                                   const float* __restrict__ denom,
                                   const float* __restrict__ feat,
                                   float* __restrict__ agg,
                                   int E, int h, int dh)
{
    const int out_d = h * dh;
    const int vecs  = out_d >> 2;
    const long long i = (long long)blockIdx.x * blockDim.x + threadIdx.x;
    if (i >= (long long)E * vecs) return;
    const int e  = (int)(i / vecs);
    const int j  = (int)(i % vecs) * 4;
    const int s  = src[e], d = dst[e];
    const int hh = j / dh;                        // dh is a multiple of 4
    const float alpha = exb[(size_t)e * h + hh] / denom[d * h + hh];
    const float4 f4 = *(const float4*)(feat + (size_t)s * out_d + j);
    float* o = agg + (size_t)d * out_d + j;
    atomicAdd(o + 0, alpha * f4.x);
    atomicAdd(o + 1, alpha * f4.y);
    atomicAdd(o + 2, alpha * f4.z);
    atomicAdd(o + 3, alpha * f4.w);
}

// ---------------------------------------------------------------------------
// Bias + optional ReLU, in place.
// ---------------------------------------------------------------------------
__global__ void gat_finalize(float* __restrict__ agg, const float* __restrict__ bias,
                             int n, int out_d, int do_relu)
{
    const int i = blockIdx.x * blockDim.x + threadIdx.x;
    if (i >= n * out_d) return;
    float v = agg[i] + bias[i % out_d];
    if (do_relu) v = fmaxf(v, 0.f);
    agg[i] = v;
}

// ---------------------------------------------------------------------------
// Host-side per-layer driver (all launches on `stream`; graph-capture safe).
// ---------------------------------------------------------------------------
static void run_gat_layer(const float* X, const float* W, const float* al, const float* ar,
                          const float* bias, _Float16* x16, _Float16* w16,
                          float* feat, float* el, float* er,
                          float* mmax, float* denom, float* exb, float* agg,
                          const int* src, const int* dst,
                          int N, int E, int in_d, int h, int dh, int do_relu,
                          hipStream_t stream)
{
    const int out_d  = h * dh;
    const int mtiles = (N + 15) / 16;

    // fp16 staging of activations and weights (both even-sized)
    int c = (N * in_d) / 2;
    cvt_f32_to_f16<<<(c + 255) / 256, 256, 0, stream>>>(X, x16, c);
    c = (out_d * in_d) / 2;
    cvt_f32_to_f16<<<(c + 255) / 256, 256, 0, stream>>>(W, w16, c);

    const dim3 gblk(32, 4);
    const int gemm_blocks = (mtiles + 3) / 4;
    if (out_d == 128)
        gat_gemm_wmma<128, 128><<<gemm_blocks, gblk, 0, stream>>>(x16, w16, feat, N);
    else
        gat_gemm_wmma<128, 64><<<gemm_blocks, gblk, 0, stream>>>(x16, w16, feat, N);

    int t = N * h;
    gat_el_er<<<(t + 255) / 256, 256, 0, stream>>>(feat, al, ar, el, er, N, h, dh);

    t = N * out_d;
    gat_init<<<(t + 255) / 256, 256, 0, stream>>>(mmax, denom, agg, N, h, out_d);

    t = E * h;
    gat_edge_max<<<(t + 255) / 256, 256, 0, stream>>>(src, dst, el, er, mmax, E, h);
    gat_edge_expsum<<<(t + 255) / 256, 256, 0, stream>>>(src, dst, el, er, mmax, exb, denom, E, h);

    const long long ta = (long long)E * (out_d / 4);
    gat_edge_aggregate<<<(unsigned)((ta + 255) / 256), 256, 0, stream>>>(
        src, dst, exb, denom, feat, agg, E, h, dh);

    t = N * out_d;
    gat_finalize<<<(t + 255) / 256, 256, 0, stream>>>(agg, bias, N, out_d, do_relu);
}

extern "C" void kernel_launch(void* const* d_in, const int* in_sizes, int n_in,
                              void* d_out, int out_size, void* d_ws, size_t ws_size,
                              hipStream_t stream)
{
    const float* feats = (const float*)d_in[0];
    const int*   src   = (const int*)d_in[1];
    const int*   dst   = (const int*)d_in[2];
    const float* W0 = (const float*)d_in[3],  *al0 = (const float*)d_in[4];
    const float* ar0 = (const float*)d_in[5], *b0  = (const float*)d_in[6];
    const float* W1 = (const float*)d_in[7],  *al1 = (const float*)d_in[8];
    const float* ar1 = (const float*)d_in[9], *b1  = (const float*)d_in[10];
    const float* W2 = (const float*)d_in[11], *al2 = (const float*)d_in[12];
    const float* ar2 = (const float*)d_in[13],*b2  = (const float*)d_in[14];

    const int D_IN = 128, H = 4, DH = 32, D_OUT = 64;
    const int N = in_sizes[0] / D_IN;   // 50000
    const int E = in_sizes[1];          // 850000

    // Workspace carve-up (~107 MB total); all chunk sizes are 32B multiples.
    float* feat  = (float*)d_ws;                 // N*128 : W-projected features
    float* act1  = feat  + (size_t)N * 128;      // N*128 : layer-0 output
    float* act2  = act1  + (size_t)N * 128;      // N*128 : layer-1 output
    float* el    = act2  + (size_t)N * 128;      // N*H
    float* er    = el    + (size_t)N * H;        // N*H
    float* mmax  = er    + (size_t)N * H;        // N*H
    float* denom = mmax  + (size_t)N * H;        // N*H
    float* exb   = denom + (size_t)N * H;        // E*H : per-edge exp values
    _Float16* x16 = (_Float16*)(exb + (size_t)E * H);   // N*128 halves
    _Float16* w16 = x16 + (size_t)N * 128;              // 128*128 halves

    run_gat_layer(feats, W0, al0, ar0, b0, x16, w16, feat, el, er, mmax, denom,
                  exb, act1, src, dst, N, E, D_IN, H, DH, /*relu=*/1, stream);
    run_gat_layer(act1, W1, al1, ar1, b1, x16, w16, feat, el, er, mmax, denom,
                  exb, act2, src, dst, N, E, H * DH, H, DH, /*relu=*/1, stream);
    // Layer 2: 1 head of 64; mean over the single head is the identity.
    run_gat_layer(act2, W2, al2, ar2, b2, x16, w16, feat, el, er, mmax, denom,
                  exb, (float*)d_out, src, dst, N, E, H * DH, 1, D_OUT, /*relu=*/0, stream);
}
